// patient_GAT_79061757985143
// MI455X (gfx1250) — compile-verified, matrix operand order checked by reading, not run on hardware
//
#include <hip/hip_runtime.h>
#include <hip/hip_bf16.h>

// ---------------- CDNA5 async global->LDS availability (probe-guarded) ----------------
#if defined(__HIP_DEVICE_COMPILE__) && defined(__has_builtin)
#if __has_builtin(__builtin_amdgcn_global_load_async_to_lds_b128) && \
    __has_builtin(__builtin_amdgcn_s_wait_asynccnt)
#define HAVE_ASYNC_LDS 1
#endif
#endif
#ifndef HAVE_ASYNC_LDS
#define HAVE_ASYNC_LDS 0
#endif

#if HAVE_ASYNC_LDS
// builtin signature (from toolchain diagnostic): (v4i AS1* src, v4i AS3* dst, imm i32, imm i32)
typedef int v4i_vs __attribute__((vector_size(16)));
typedef __attribute__((address_space(1))) v4i_vs* gv4i_p;   // global
typedef __attribute__((address_space(3))) v4i_vs* lv4i_p;   // LDS
#endif

// ---------------- types ----------------
typedef __bf16 v16bf __attribute__((ext_vector_type(16)));
typedef __bf16 v8bf  __attribute__((ext_vector_type(8)));
typedef float  v8f   __attribute__((ext_vector_type(8)));

static __device__ __forceinline__ __bf16 f2bf(float f) {
    unsigned u = __float_as_uint(f);
    unsigned r = (u + 0x7FFFu + ((u >> 16) & 1u)) >> 16;   // RNE
    unsigned short s = (unsigned short)r;
    __bf16 b;
    __builtin_memcpy(&b, &s, 2);
    return b;
}

static __device__ __forceinline__ void atomicMaxF(float* addr, float val) {
    if (val >= 0.0f) atomicMax((int*)addr, __float_as_int(val));
    else             atomicMin((unsigned int*)addr, __float_as_uint(val));
}

// ---------------- utility kernels ----------------
__global__ void __launch_bounds__(256) k_fill_f32(float* p, long long n, float v) {
    long long t = (long long)blockIdx.x * 256 + threadIdx.x;
    if (t < n) p[t] = v;
}

__global__ void __launch_bounds__(256) k_f32_to_bf16(const float* __restrict__ in,
                                                     __bf16* __restrict__ out, long long n) {
    long long t = (long long)blockIdx.x * 256 + threadIdx.x;
    if (t < n) out[t] = f2bf(in[t]);
}

// W[K,HC] (row-major f32) -> Bt[HC,K] (row-major bf16). Tiny, done once; result lives in L2.
__global__ void __launch_bounds__(256) k_w_transpose_bf16(const float* __restrict__ W,
                                                          __bf16* __restrict__ out,
                                                          int K, int HC) {
    long long t = (long long)blockIdx.x * 256 + threadIdx.x;
    if (t >= (long long)K * HC) return;
    const int n = (int)(t / K), k = (int)(t % K);
    out[t] = f2bf(W[(size_t)k * HC + n]);
}

// ---------------- WMMA GEMM: C[M,256] = A[M,K](bf16) @ B[K,256](bf16), fp32 accum ----------------
// block: 256 threads = 8 waves; block tile 64(M) x 256(N); wave w: rows (w&3)*16, cols (w>>2)*128.
// A: double-buffered LDS tiles staged with async global->LDS (ASYNCcnt pipelined).
// B: pre-transposed Bt[256][K] read as fragments straight from L2 (weights are L2-resident).
__global__ void __launch_bounds__(256) k_gemm_bf16_wmma(const __bf16* __restrict__ A,
                                                        const __bf16* __restrict__ Bt,
                                                        float* __restrict__ C,
                                                        int M, int K) {
    __shared__ __attribute__((aligned(16))) __bf16 sA[2][64 * 32];

    const int tid   = threadIdx.x;
    const int m0    = blockIdx.x * 64;
    const int wave  = tid >> 5;
    const int lane  = tid & 31;
    const int mbase = (wave & 3) * 16;
    const int nbase = (wave >> 2) * 128;
    const int kh    = lane >> 4;      // K-half owned by this lane (16-bit A/B fragment layout)
    const int l16   = lane & 15;
    const bool fullTile = (m0 + 64 <= M);

    const int srow = tid >> 2;                 // staging row 0..63 (one 16B segment per thread)
    const int sseg = (tid & 3) * 8;
    const __bf16* aRow = A + (size_t)(m0 + srow) * K + sseg;

    v8f acc[8] = {};

    auto stage = [&](int k0, int b) {
#if HAVE_ASYNC_LDS
        if (fullTile) {
            __builtin_amdgcn_global_load_async_to_lds_b128(
                (gv4i_p)(const_cast<__bf16*>(aRow + k0)),
                (lv4i_p)(&sA[b][srow * 32 + sseg]),
                0, 0);
            return;
        }
#endif
        v8bf v = {};
        if (m0 + srow < M) v = *(const v8bf*)(aRow + k0);
        *(v8bf*)(&sA[b][srow * 32 + sseg]) = v;
    };

    int buf = 0;
    stage(0, 0);
    for (int k0 = 0; k0 < K; k0 += 32) {
        if (k0 + 32 < K) {
            stage(k0 + 32, buf ^ 1);           // tile i+1 in flight while tile i is consumed
#if HAVE_ASYNC_LDS
            if (fullTile) __builtin_amdgcn_s_wait_asynccnt(1);   // tile i has landed
#endif
        }
#if HAVE_ASYNC_LDS
        else if (fullTile) __builtin_amdgcn_s_wait_asynccnt(0);
#endif
        __syncthreads();

        // A fragment: lane (kh,l16) holds row mbase+l16, K = kh*8..+7 and kh*8+16..+23
        v8bf alo = *(const v8bf*)(&sA[buf][(mbase + l16) * 32 + kh * 8]);
        v8bf ahi = *(const v8bf*)(&sA[buf][(mbase + l16) * 32 + kh * 8 + 16]);
        v16bf av = __builtin_shufflevector(alo, ahi,
                                           0,1,2,3,4,5,6,7,8,9,10,11,12,13,14,15);

        #pragma unroll
        for (int t = 0; t < 8; ++t) {
            const __bf16* bp = Bt + (size_t)(nbase + t * 16 + l16) * K + k0 + kh * 8;
            v8bf blo = *(const v8bf*)bp;
            v8bf bhi = *(const v8bf*)(bp + 16);
            v16bf bv = __builtin_shufflevector(blo, bhi,
                                               0,1,2,3,4,5,6,7,8,9,10,11,12,13,14,15);
            acc[t] = __builtin_amdgcn_wmma_f32_16x16x32_bf16(
                         false, av, false, bv, (short)0, acc[t], false, false);
        }
        __syncthreads();                       // all waves done with sA[buf] before it is re-staged
        buf ^= 1;
    }

    // epilogue: D layout -> VGPR v: lanes 0-15 row M=v col=lane; lanes 16-31 row M=v+8 col=lane-16
    #pragma unroll
    for (int t = 0; t < 8; ++t) {
        const int col = nbase + t * 16 + l16;
        #pragma unroll
        for (int v = 0; v < 8; ++v) {
            const int row = m0 + mbase + kh * 8 + v;
            if (row < M) C[(size_t)row * 256 + col] = acc[t][v];
        }
    }
}

// ---------------- per-node attention logits: e_src/e_dst [N,4] ----------------
__global__ void __launch_bounds__(256) k_node_dots(const float* __restrict__ h,
                                                   const float* __restrict__ aS,
                                                   const float* __restrict__ aD,
                                                   float* __restrict__ eS,
                                                   float* __restrict__ eD, int N) {
    const long long gt = (long long)blockIdx.x * 256 + threadIdx.x;
    const int n = (int)(gt >> 5), lane = (int)(gt & 31);
    if (n >= N) return;
    const float* hp = h + (size_t)n * 256 + lane * 8;
    float pa = 0.f, pb = 0.f;
    #pragma unroll
    for (int i = 0; i < 8; ++i) {
        const float v = hp[i];
        pa += v * aS[lane * 8 + i];
        pb += v * aD[lane * 8 + i];
    }
    #pragma unroll
    for (int m = 1; m <= 4; m <<= 1) {   // reduce across the 8 lanes of each head (wave32 shuffles)
        pa += __shfl_xor(pa, m, 32);
        pb += __shfl_xor(pb, m, 32);
    }
    if ((lane & 7) == 0) {
        const int hd = lane >> 3;
        eS[(size_t)n * 4 + hd] = pa;
        eD[(size_t)n * 4 + hd] = pb;
    }
}

static __device__ __forceinline__ void edge_ids(const long long* ei, int E, int e, int& s, int& d) {
    if (e < E) { s = (int)ei[e]; d = (int)ei[(size_t)E + e]; }
    else       { s = d = e - E; }   // self-loops appended
}

// ---------------- softmax pass 1: segment max ----------------
__global__ void __launch_bounds__(256) k_edge_max(const long long* __restrict__ ei,
                                                  const float* __restrict__ eS,
                                                  const float* __restrict__ eD,
                                                  float* __restrict__ mmax,
                                                  int E, int Etot) {
    const long long t = (long long)blockIdx.x * 256 + threadIdx.x;
    if (t >= (long long)Etot * 4) return;
    const int e = (int)(t >> 2), hd = (int)(t & 3);
    int s, d; edge_ids(ei, E, e, s, d);
    float v = eS[(size_t)s * 4 + hd] + eD[(size_t)d * 4 + hd];
    v = v > 0.f ? v : 0.2f * v;                 // leaky relu
    atomicMaxF(&mmax[(size_t)d * 4 + hd], v);
}

// ---------------- softmax pass 2: exp + segment sum ----------------
__global__ void __launch_bounds__(256) k_edge_exp(const long long* __restrict__ ei,
                                                  const float* __restrict__ eS,
                                                  const float* __restrict__ eD,
                                                  const float* __restrict__ mmax,
                                                  float* __restrict__ exb,
                                                  float* __restrict__ ssum,
                                                  int E, int Etot) {
    const long long t = (long long)blockIdx.x * 256 + threadIdx.x;
    if (t >= (long long)Etot * 4) return;
    const int e = (int)(t >> 2), hd = (int)(t & 3);
    int s, d; edge_ids(ei, E, e, s, d);
    float v = eS[(size_t)s * 4 + hd] + eD[(size_t)d * 4 + hd];
    v = v > 0.f ? v : 0.2f * v;
    const float ex = __expf(v - mmax[(size_t)d * 4 + hd]);
    exb[(size_t)e * 4 + hd] = ex;
    atomicAdd(&ssum[(size_t)d * 4 + hd], ex);
}

// ---------------- softmax pass 3: normalize + scatter messages ----------------
// one wave per edge: lane handles 8 channels (head = lane>>3); 32B vector gather, atomic scatter
__global__ void __launch_bounds__(256) k_edge_scatter(const long long* __restrict__ ei,
                                                      const float* __restrict__ h,
                                                      const float* __restrict__ exb,
                                                      const float* __restrict__ ssum,
                                                      float* __restrict__ agg,
                                                      int E, int Etot) {
    const long long t = (long long)blockIdx.x * 256 + threadIdx.x;
    const int e = (int)(t >> 5), lane = (int)(t & 31);
    if (e >= Etot) return;
    int s, d; edge_ids(ei, E, e, s, d);
    const int hd = lane >> 3;
    const float alpha = exb[(size_t)e * 4 + hd] / ssum[(size_t)d * 4 + hd];
    const float4* hs = (const float4*)(h + (size_t)s * 256 + lane * 8);
    const float4 a = hs[0], b = hs[1];
    float* op = agg + (size_t)d * 256 + lane * 8;
    atomicAdd(op + 0, a.x * alpha); atomicAdd(op + 1, a.y * alpha);
    atomicAdd(op + 2, a.z * alpha); atomicAdd(op + 3, a.w * alpha);
    atomicAdd(op + 4, b.x * alpha); atomicAdd(op + 5, b.y * alpha);
    atomicAdd(op + 6, b.z * alpha); atomicAdd(op + 7, b.w * alpha);
}

// ---------------- bias + relu (+ optional fp32 out / bf16 out for next layer) ----------------
__global__ void __launch_bounds__(256) k_bias_relu(const float* __restrict__ agg,
                                                   const float* __restrict__ bias,
                                                   float* __restrict__ fout,
                                                   __bf16* __restrict__ bout,
                                                   long long total) {
    const long long t = (long long)blockIdx.x * 256 + threadIdx.x;
    if (t >= total) return;
    const int ch = (int)(t & 255);
    float v = agg[t] + bias[ch];
    v = v > 0.f ? v : 0.f;
    if (fout) fout[t] = v;
    if (bout) bout[t] = f2bf(v);
}

// ---------------- pooling ----------------
__global__ void __launch_bounds__(256) k_pool_sum(const float* __restrict__ act,
                                                  const long long* __restrict__ batch,
                                                  float* __restrict__ pooled, long long total) {
    const long long t = (long long)blockIdx.x * 256 + threadIdx.x;
    if (t >= total) return;
    const int n = (int)(t >> 8), ch = (int)(t & 255);
    const int g = (int)batch[n];
    atomicAdd(&pooled[(size_t)g * 256 + ch], act[t]);
}

__global__ void __launch_bounds__(256) k_pool_cnt(const long long* __restrict__ batch,
                                                  float* __restrict__ cnt, int N) {
    const int t = blockIdx.x * 256 + threadIdx.x;
    if (t >= N) return;
    atomicAdd(&cnt[(int)batch[t]], 1.0f);
}

// ---------------- final linear: out[G,16] = (pooled/cnt) @ Wlin + blin ----------------
__global__ void __launch_bounds__(256) k_final(const float* __restrict__ pooled,
                                               const float* __restrict__ cnt,
                                               const float* __restrict__ Wlin,
                                               const float* __restrict__ blin,
                                               float* __restrict__ out, int G, int OUT) {
    const int t = blockIdx.x * 256 + threadIdx.x;
    if (t >= G * OUT) return;
    const int g = t / OUT, o = t % OUT;
    const float c = cnt[g];
    const float denom = c > 1.f ? c : 1.f;
    float s = 0.f;
    for (int j = 0; j < 256; ++j)
        s += pooled[(size_t)g * 256 + j] * Wlin[(size_t)j * OUT + o];
    out[t] = s / denom + blin[o];
}

// ---------------- host ----------------
static inline long long cdiv(long long a, long long b) { return (a + b - 1) / b; }

extern "C" void kernel_launch(void* const* d_in, const int* in_sizes, int n_in,
                              void* d_out, int out_size, void* d_ws, size_t ws_size,
                              hipStream_t stream) {
    const float*     x    = (const float*)d_in[0];
    const long long* ei   = (const long long*)d_in[1];
    const long long* bat  = (const long long*)d_in[2];
    const float*     W1   = (const float*)d_in[3];
    const float*     aS1  = (const float*)d_in[4];
    const float*     aD1  = (const float*)d_in[5];
    const float*     b1   = (const float*)d_in[6];
    const float*     W2   = (const float*)d_in[7];
    const float*     aS2  = (const float*)d_in[8];
    const float*     aD2  = (const float*)d_in[9];
    const float*     b2   = (const float*)d_in[10];
    const float*     Wlin = (const float*)d_in[11];
    const float*     blin = (const float*)d_in[12];
    float*           out  = (float*)d_out;

    const int N    = in_sizes[2];            // batch has N entries
    const int E    = in_sizes[1] / 2;        // edge_index [2,E]
    const int FIN  = in_sizes[0] / N;        // 128
    const int HC   = in_sizes[3] / FIN;      // 256
    const int OUT  = in_sizes[11] / HC;      // 16
    const int G    = out_size / OUT;         // 64
    const int Etot = E + N;                  // with self-loops

    // carve workspace (256B aligned chunks)
    size_t off = 0;
    auto wsp = [&](size_t bytes) -> void* {
        void* p = (char*)d_ws + off;
        off += (bytes + 255) & ~(size_t)255;
        return p;
    };
    __bf16* xbf    = (__bf16*)wsp((size_t)N * FIN * 2);
    __bf16* w1bt   = (__bf16*)wsp((size_t)FIN * HC * 2);   // transposed [HC][FIN]
    __bf16* w2bt   = (__bf16*)wsp((size_t)HC * HC * 2);    // transposed [HC][HC]
    __bf16* hbf    = (__bf16*)wsp((size_t)N * HC * 2);
    float*  hbuf   = (float*)wsp((size_t)N * HC * 4);
    float*  agg    = (float*)wsp((size_t)N * HC * 4);
    float*  eS     = (float*)wsp((size_t)N * 4 * 4);
    float*  eD     = (float*)wsp((size_t)N * 4 * 4);
    float*  mmax   = (float*)wsp((size_t)N * 4 * 4);
    float*  ssum   = (float*)wsp((size_t)N * 4 * 4);
    float*  exb    = (float*)wsp((size_t)Etot * 4 * 4);
    float*  pooled = (float*)wsp((size_t)G * HC * 4);
    float*  cnt    = (float*)wsp((size_t)G * 4);
    (void)ws_size; (void)n_in;

    const dim3 B256(256);
    const long long nHC = (long long)N * HC;

    // precision conversion / weight transposition for WMMA GEMMs
    k_f32_to_bf16<<<cdiv((long long)N * FIN, 256), B256, 0, stream>>>(x, xbf, (long long)N * FIN);
    k_w_transpose_bf16<<<cdiv((long long)FIN * HC, 256), B256, 0, stream>>>(W1, w1bt, FIN, HC);
    k_w_transpose_bf16<<<cdiv((long long)HC * HC, 256),  B256, 0, stream>>>(W2, w2bt, HC, HC);

    const int gemmBlocks = (int)cdiv(N, 64);
    const int edgeHGrid  = (int)cdiv((long long)Etot * 4, 256);
    const int edgeWGrid  = (int)cdiv((long long)Etot * 32, 256);
    const int nodeWGrid  = (int)cdiv((long long)N * 32, 256);
    const int nchGrid    = (int)cdiv(nHC, 256);

    // ---------- layer 1 ----------
    k_gemm_bf16_wmma<<<gemmBlocks, B256, 0, stream>>>(xbf, w1bt, hbuf, N, FIN);
    k_node_dots<<<nodeWGrid, B256, 0, stream>>>(hbuf, aS1, aD1, eS, eD, N);
    k_fill_f32<<<cdiv((long long)N * 4, 256), B256, 0, stream>>>(mmax, (long long)N * 4, -__builtin_huge_valf());
    k_fill_f32<<<cdiv((long long)N * 4, 256), B256, 0, stream>>>(ssum, (long long)N * 4, 0.f);
    k_fill_f32<<<nchGrid, B256, 0, stream>>>(agg, nHC, 0.f);
    k_edge_max<<<edgeHGrid, B256, 0, stream>>>(ei, eS, eD, mmax, E, Etot);
    k_edge_exp<<<edgeHGrid, B256, 0, stream>>>(ei, eS, eD, mmax, exb, ssum, E, Etot);
    k_edge_scatter<<<edgeWGrid, B256, 0, stream>>>(ei, hbuf, exb, ssum, agg, E, Etot);
    k_bias_relu<<<nchGrid, B256, 0, stream>>>(agg, b1, (float*)nullptr, hbf, nHC);

    // ---------- layer 2 ----------
    k_gemm_bf16_wmma<<<gemmBlocks, B256, 0, stream>>>(hbf, w2bt, hbuf, N, HC);
    k_node_dots<<<nodeWGrid, B256, 0, stream>>>(hbuf, aS2, aD2, eS, eD, N);
    k_fill_f32<<<cdiv((long long)N * 4, 256), B256, 0, stream>>>(mmax, (long long)N * 4, -__builtin_huge_valf());
    k_fill_f32<<<cdiv((long long)N * 4, 256), B256, 0, stream>>>(ssum, (long long)N * 4, 0.f);
    k_fill_f32<<<nchGrid, B256, 0, stream>>>(agg, nHC, 0.f);
    k_edge_max<<<edgeHGrid, B256, 0, stream>>>(ei, eS, eD, mmax, E, Etot);
    k_edge_exp<<<edgeHGrid, B256, 0, stream>>>(ei, eS, eD, mmax, exb, ssum, E, Etot);
    k_edge_scatter<<<edgeWGrid, B256, 0, stream>>>(ei, hbuf, exb, ssum, agg, E, Etot);
    k_bias_relu<<<nchGrid, B256, 0, stream>>>(agg, b2, agg, (__bf16*)nullptr, nHC);

    // ---------- pooling + final linear ----------
    k_fill_f32<<<cdiv((long long)G * HC, 256), B256, 0, stream>>>(pooled, (long long)G * HC, 0.f);
    k_fill_f32<<<1, B256, 0, stream>>>(cnt, (long long)G, 0.f);
    k_pool_sum<<<nchGrid, B256, 0, stream>>>(agg, bat, pooled, nHC);
    k_pool_cnt<<<(int)cdiv(N, 256), B256, 0, stream>>>(bat, cnt, N);
    k_final<<<(int)cdiv((long long)G * OUT, 256), B256, 0, stream>>>(pooled, cnt, Wlin, blin, out, G, OUT);
}
